// MultiheadSelfAttention_32392643346760
// MI455X (gfx1250) — matrix operand order repaired and checked by reading, not compiled, not run on hardware
//
#include <hip/hip_runtime.h>
#include <hip/hip_bf16.h>
#include <math.h>

// ---------------------------------------------------------------------------
// MI455X (gfx1250) multi-head self-attention with RoPE, causal, flash-style.
//   B=2, T=2048, D=1024, H=16, dh=64.  fp32 in/out, bf16 WMMA internally
//   (V_WMMA_F32_16X16X32_BF16, fp32 accumulate).
//   Round 2: one-shot fp32->bf16 pre-pass for x and all weights so the GEMM
//   inner loops are pure b128 bf16 loads + WMMA (no per-K-step conversions).
// ---------------------------------------------------------------------------

typedef __attribute__((ext_vector_type(16))) __bf16 v16bf;
typedef __attribute__((ext_vector_type(8)))  float  v8f;

static constexpr int NH     = 16;
static constexpr int DH     = 64;
static constexpr int DMODEL = 1024;
static constexpr int TSEQ   = 2048;
static constexpr int BATCH  = 2;

// ---- WMMA fragment loaders (CDNA5 16x16x32 bf16 layouts, wave32) ----------

// A-frag (16x32, MxK): lane L -> row L&15; K runs kb..kb+15,
// kb = (L<16)?0:16.  One 32-byte contiguous run per lane.
__device__ inline v16bf a_frag_from_bf16(const __bf16* src, int ld, int lane) {
  const int m  = lane & 15;
  const int kb = (lane < 16) ? 0 : 16;
  const __bf16* p = src + (size_t)m * ld + kb;
  v16bf a;
#pragma unroll
  for (int e = 0; e < 8; ++e) a[e]     = p[e];
#pragma unroll
  for (int e = 0; e < 8; ++e) a[e + 8] = p[e + 8];
  return a;
}

// B-frag (32x16, KxN): lane L -> column L&15; K run kb16+e, kb16=(L<16)?0:16.
// One 32-byte contiguous run per lane.
__device__ inline v16bf b_frag_from_bf16row(const __bf16* row, int lane) {
  const int kb = (lane < 16) ? 0 : 16;
  v16bf b;
#pragma unroll
  for (int e = 0; e < 16; ++e) b[e] = row[kb + e];
  return b;
}

// ---------------------------------------------------------------------------
// Kernel 0: one-shot fp32 -> bf16 conversion (x and the four weight mats).
// ---------------------------------------------------------------------------
__global__ void f32_to_bf16_kernel(const float* __restrict__ src,
                                   __bf16* __restrict__ dst, int n) {
  const int i = (blockIdx.x * blockDim.x + threadIdx.x) * 4;
  if (i < n) {
    const float4 v = *(const float4*)(src + i);
    dst[i + 0] = (__bf16)v.x;
    dst[i + 1] = (__bf16)v.y;
    dst[i + 2] = (__bf16)v.z;
    dst[i + 3] = (__bf16)v.w;
  }
}

// ---------------------------------------------------------------------------
// Kernel 1: Q/K/V projection (out = x @ W^T) + RoPE on Q,K.
//   grid = (BT/16, NH, 3), block = 32 (one wave computes a 16x64 tile = one
//   head's dh for 16 tokens; A-frag reused across 4 WMMAs per K-step).
//   Q,K stored bf16 row-major [b,h,t,dh]; V stored transposed [b,h,dh,t].
// ---------------------------------------------------------------------------
__global__ void qkv_rope_kernel(const __bf16* __restrict__ xbf,
                                const __bf16* __restrict__ Wqbf,
                                const __bf16* __restrict__ Wkbf,
                                const __bf16* __restrict__ Wvbf,
                                const int*    __restrict__ tpos,
                                __bf16* __restrict__ Qbf,
                                __bf16* __restrict__ Kbf,
                                __bf16* __restrict__ Vt) {
  const int lane  = threadIdx.x;
  const int mt    = blockIdx.x;          // 16-row tile of B*T
  const int head  = blockIdx.y;          // 64 output columns
  const int which = blockIdx.z;          // 0=Q 1=K 2=V
  const __bf16* W = (which == 0) ? Wqbf : (which == 1) ? Wkbf : Wvbf;

  const int row0 = mt * 16;
  const int col0 = head * DH;
  const int n15  = lane & 15;
  const int mloc = (lane < 16) ? 0 : 8;

  v8f acc[4] = {};                       // four 16x16 column tiles

  for (int k0 = 0; k0 < DMODEL; k0 += 32) {
    __builtin_prefetch(xbf + (size_t)row0 * DMODEL + k0 + 64, 0, 0);
    v16bf a = a_frag_from_bf16(xbf + (size_t)row0 * DMODEL + k0, DMODEL, lane);
#pragma unroll
    for (int j = 0; j < 4; ++j) {
      const int n = col0 + j * 16 + n15;       // weight row == B-frag column
      v16bf b = b_frag_from_bf16row(W + (size_t)n * DMODEL + k0, lane);
      acc[j] = __builtin_amdgcn_wmma_f32_16x16x32_bf16(
          false, a, false, b, (short)0, acc[j], false, false);
    }
  }

  if (which == 2) {
    // V transposed: Vt[(bh*DH + d)*T + t]; element index r walks t -> contiguous
#pragma unroll
    for (int j = 0; j < 4; ++j) {
      const int d = j * 16 + n15;
#pragma unroll
      for (int r = 0; r < 8; ++r) {
        const int gm = row0 + r + mloc;
        const int b = gm / TSEQ, t = gm % TSEQ;
        Vt[((size_t)(b * NH + head) * DH + d) * TSEQ + t] = (__bf16)acc[j][r];
      }
    }
  } else {
    __bf16* dst = (which == 0) ? Qbf : Kbf;
#pragma unroll
    for (int j = 0; j < 4; ++j) {
      const int d = j * 16 + n15;                       // 0..63 inside head
      const float inv_freq = __powf(10000.0f, -(float)(d >> 1) * (1.0f / 32.0f));
      const bool even = (d & 1) == 0;
#pragma unroll
      for (int r = 0; r < 8; ++r) {
        const float v  = acc[j][r];
        const float vp = __shfl_xor(v, 1, 32);          // paired column (lane^1)
        const int gm = row0 + r + mloc;
        const int b = gm / TSEQ, t = gm % TSEQ;
        const float pos = (float)tpos[(size_t)b * TSEQ + t];
        float s, c;
        __sincosf(pos * inv_freq, &s, &c);
        const float o = v * c + (even ? -vp * s : vp * s);
        dst[((size_t)(b * NH + head) * TSEQ + t) * DH + d] = (__bf16)o;
      }
    }
  }
}

// ---------------------------------------------------------------------------
// Kernel 2: flash attention per (b, h, 16-query tile).  block = 32 (1 wave).
//   Per 32-key step: 4 WMMAs for S = Q*K^T (K=64 in 2 chunks x 2 key tiles),
//   online softmax through LDS (row transpose), 4 WMMAs for O += P*V.
//   Q-frags loaded once and reused across the whole key loop.
// ---------------------------------------------------------------------------
__global__ void flash_attn_kernel(const __bf16* __restrict__ Qbf,
                                  const __bf16* __restrict__ Kbf,
                                  const __bf16* __restrict__ Vt,
                                  __bf16* __restrict__ Obf) {
  __shared__ float Sl[16][33];
  __shared__ float alpha[16];
  __shared__ float lsum[16];

  const int lane = threadIdx.x;
  const int q0   = blockIdx.x * 16;
  const int h    = blockIdx.y;
  const int b    = blockIdx.z;
  const int bh   = b * NH + h;

  const __bf16* Qbase = Qbf + ((size_t)bh * TSEQ + q0) * DH;
  const __bf16* Kbase = Kbf + (size_t)bh * TSEQ * DH;
  const __bf16* Vbase = Vt  + (size_t)bh * DH * TSEQ;

  const int n15  = lane & 15;
  const int mloc = (lane < 16) ? 0 : 8;
  const int kb16 = (lane < 16) ? 0 : 16;
  const int kbA  = (lane < 16) ? 0 : 16;

  const v16bf aq0 = a_frag_from_bf16(Qbase +  0, DH, lane);  // features 0..31
  const v16bf aq1 = a_frag_from_bf16(Qbase + 32, DH, lane);  // features 32..63

  v8f o[4] = {};
  float mrun = -INFINITY, lrun = 0.0f;     // lanes 0..15 own row = lane

  for (int kt0 = 0; kt0 <= q0 + 15; kt0 += 32) {
    // ---- S = (Q K^T) * 1/sqrt(dh), two 16-key tiles ----
    v8f s0 = {}, s1 = {};
    const __bf16* krow0 = Kbase + (size_t)(kt0 + n15) * DH;
    const __bf16* krow1 = Kbase + (size_t)(kt0 + 16 + n15) * DH;
    {
      v16bf b00 = b_frag_from_bf16row(krow0 +  0, lane);
      v16bf b10 = b_frag_from_bf16row(krow1 +  0, lane);
      s0 = __builtin_amdgcn_wmma_f32_16x16x32_bf16(false, aq0, false, b00, (short)0, s0, false, false);
      s1 = __builtin_amdgcn_wmma_f32_16x16x32_bf16(false, aq0, false, b10, (short)0, s1, false, false);
      v16bf b01 = b_frag_from_bf16row(krow0 + 32, lane);
      v16bf b11 = b_frag_from_bf16row(krow1 + 32, lane);
      s0 = __builtin_amdgcn_wmma_f32_16x16x32_bf16(false, aq1, false, b01, (short)0, s0, false, false);
      s1 = __builtin_amdgcn_wmma_f32_16x16x32_bf16(false, aq1, false, b11, (short)0, s1, false, false);
    }
    // ---- causal mask + scale, scatter to LDS (row-major) ----
#pragma unroll
    for (int r = 0; r < 8; ++r) {
      const int q = q0 + r + mloc;
      float v0 = s0[r] * 0.125f;
      float v1 = s1[r] * 0.125f;
      if (kt0 + n15 > q)      v0 = -INFINITY;
      if (kt0 + 16 + n15 > q) v1 = -INFINITY;
      Sl[r + mloc][n15]      = v0;
      Sl[r + mloc][16 + n15] = v1;
    }
    __syncthreads();
    // ---- online softmax: lanes 0..15 each own one query row ----
    if (lane < 16) {
      float tm = -INFINITY;
#pragma unroll
      for (int j = 0; j < 32; ++j) tm = fmaxf(tm, Sl[lane][j]);
      const float mnew = fmaxf(mrun, tm);
      const float corr = __expf(mrun - mnew);
      float ts = 0.0f;
#pragma unroll
      for (int j = 0; j < 32; ++j) {
        const float p = __expf(Sl[lane][j] - mnew);
        Sl[lane][j] = p;
        ts += p;
      }
      lrun = lrun * corr + ts;
      mrun = mnew;
      alpha[lane] = corr;
    }
    __syncthreads();
    // ---- rescale O, then O += P * V ----
#pragma unroll
    for (int r = 0; r < 8; ++r) {
      const float am = alpha[r + mloc];
#pragma unroll
      for (int j = 0; j < 4; ++j) o[j][r] *= am;
    }
    v16bf ap;                                   // P as A-frag (row = n15)
#pragma unroll
    for (int e = 0; e < 8; ++e) ap[e]     = (__bf16)Sl[n15][kbA + e];
#pragma unroll
    for (int e = 0; e < 8; ++e) ap[e + 8] = (__bf16)Sl[n15][kbA + 8 + e];
#pragma unroll
    for (int j = 0; j < 4; ++j) {
      const __bf16* vrow = Vbase + (size_t)(j * 16 + n15) * TSEQ + kt0 + kb16;
      v16bf bv;
#pragma unroll
      for (int e = 0; e < 16; ++e) bv[e] = vrow[e];   // contiguous along keys
      o[j] = __builtin_amdgcn_wmma_f32_16x16x32_bf16(
          false, ap, false, bv, (short)0, o[j], false, false);
    }
    __syncthreads();                             // Sl reused next iteration
  }

  if (lane < 16) lsum[lane] = lrun;
  __syncthreads();
  // ---- normalize and emit bf16 attention output, layout [b*T+t][h*64+d] ----
#pragma unroll
  for (int r = 0; r < 8; ++r) {
    const int t = q0 + r + mloc;
    const float inv = 1.0f / lsum[r + mloc];
#pragma unroll
    for (int j = 0; j < 4; ++j) {
      const int d = j * 16 + n15;
      Obf[(size_t)(b * TSEQ + t) * DMODEL + h * DH + d] = (__bf16)(o[j][r] * inv);
    }
  }
}

// ---------------------------------------------------------------------------
// Kernel 3: out = attnOut @ Wo^T  (fp32 result).  grid = (BT/16, D/64).
// ---------------------------------------------------------------------------
__global__ void out_proj_kernel(const __bf16* __restrict__ Obf,
                                const __bf16* __restrict__ Wobf,
                                float* __restrict__ out) {
  const int lane = threadIdx.x;
  const int row0 = blockIdx.x * 16;
  const int col0 = blockIdx.y * 64;
  const int n15  = lane & 15;
  const int mloc = (lane < 16) ? 0 : 8;

  v8f acc[4] = {};
  for (int k0 = 0; k0 < DMODEL; k0 += 32) {
    __builtin_prefetch(Obf + (size_t)row0 * DMODEL + k0 + 64, 0, 0);
    v16bf a = a_frag_from_bf16(Obf + (size_t)row0 * DMODEL + k0, DMODEL, lane);
#pragma unroll
    for (int j = 0; j < 4; ++j) {
      const int n = col0 + j * 16 + n15;
      v16bf b = b_frag_from_bf16row(Wobf + (size_t)n * DMODEL + k0, lane);
      acc[j] = __builtin_amdgcn_wmma_f32_16x16x32_bf16(
          false, a, false, b, (short)0, acc[j], false, false);
    }
  }
#pragma unroll
  for (int j = 0; j < 4; ++j)
#pragma unroll
    for (int r = 0; r < 8; ++r)
      out[(size_t)(row0 + r + mloc) * DMODEL + col0 + j * 16 + n15] = acc[j][r];
}

// ---------------------------------------------------------------------------
extern "C" void kernel_launch(void* const* d_in, const int* in_sizes, int n_in,
                              void* d_out, int out_size, void* d_ws, size_t ws_size,
                              hipStream_t stream) {
  const float* x    = (const float*)d_in[0];
  const float* Wq   = (const float*)d_in[1];
  const float* Wk   = (const float*)d_in[2];
  const float* Wv   = (const float*)d_in[3];
  const float* Wo   = (const float*)d_in[4];
  const int*   tpos = (const int*)d_in[5];

  const size_t NELT = (size_t)BATCH * NH * TSEQ * DH;   // 4M elements
  const size_t WELT = (size_t)DMODEL * DMODEL;          // 1M elements

  __bf16* Qbf  = (__bf16*)d_ws;                         // 8 MB each
  __bf16* Kbf  = Qbf + NELT;
  __bf16* Vt   = Kbf + NELT;
  __bf16* Obf  = Vt  + NELT;
  __bf16* xbf  = Obf + NELT;                            // 8 MB
  __bf16* Wqbf = xbf + NELT;                            // 2 MB each
  __bf16* Wkbf = Wqbf + WELT;
  __bf16* Wvbf = Wkbf + WELT;
  __bf16* Wobf = Wvbf + WELT;                           // total 48 MB

  // one-shot operand down-conversion (removes all cvt from GEMM inner loops)
  f32_to_bf16_kernel<<<(int)(NELT / 4 / 256), 256, 0, stream>>>(x,  xbf,  (int)NELT);
  f32_to_bf16_kernel<<<(int)(WELT / 4 / 256), 256, 0, stream>>>(Wq, Wqbf, (int)WELT);
  f32_to_bf16_kernel<<<(int)(WELT / 4 / 256), 256, 0, stream>>>(Wk, Wkbf, (int)WELT);
  f32_to_bf16_kernel<<<(int)(WELT / 4 / 256), 256, 0, stream>>>(Wv, Wvbf, (int)WELT);
  f32_to_bf16_kernel<<<(int)(WELT / 4 / 256), 256, 0, stream>>>(Wo, Wobf, (int)WELT);

  qkv_rope_kernel<<<dim3((BATCH * TSEQ) / 16, NH, 3), 32, 0, stream>>>(
      xbf, Wqbf, Wkbf, Wvbf, tpos, Qbf, Kbf, Vt);
  flash_attn_kernel<<<dim3(TSEQ / 16, NH, BATCH), 32, 0, stream>>>(
      Qbf, Kbf, Vt, Obf);
  out_proj_kernel<<<dim3((BATCH * TSEQ) / 16, DMODEL / 64, 1), 32, 0, stream>>>(
      Obf, Wobf, (float*)d_out);
}
